// FusedLlamaLayerIntra_80908593922049
// MI455X (gfx1250) — matrix operand order repaired and checked by reading, not compiled, not run on hardware
//
#include <hip/hip_runtime.h>
#include <hip/hip_bf16.h>
#include <stdint.h>

// ---------------------------------------------------------------------------
// Fused LLaMA layer for MI455X (gfx1250, wave32, WMMA).
// LoRA folded into bf16 effective weights once; all GEMMs + flash attention
// run on v_wmma_f32_16x16x32_bf16 with f32 accumulation.
// ---------------------------------------------------------------------------

typedef __bf16 bf16_t;
typedef __attribute__((ext_vector_type(16))) __bf16 v16bf;
typedef __attribute__((ext_vector_type(8)))  __bf16 v8bf;
typedef __attribute__((ext_vector_type(8)))  float  v8f;

#define WMMA_BF16(A,B,C) \
  __builtin_amdgcn_wmma_f32_16x16x32_bf16(false,(A),false,(B),(short)0,(C),false,false)

enum { Bc=2, Sc=2048, Dc=2048, Hc=16, HDc=128, FFc=5632, Rc=16, Mc=Bc*Sc };

// ---------------------------------------------------------------------------
// 1) Fold LoRA into weights and cast to bf16:  we[n,k] = W[n,k] + lb[n,:]·la[:,k]
// ---------------------------------------------------------------------------
__global__ __launch_bounds__(256) void lora_fold(
    const float* __restrict__ Wf, const float* __restrict__ la,
    const float* __restrict__ lb, bf16_t* __restrict__ we,
    int Nn, int Kk)
{
  size_t i = (size_t)blockIdx.x * 256 + threadIdx.x;
  size_t total = (size_t)Nn * Kk;
  if (i >= total) return;
  int n = (int)(i / Kk);
  int k = (int)(i % Kk);
  float acc = Wf[i];
  #pragma unroll
  for (int r = 0; r < Rc; ++r)
    acc += lb[(size_t)n * Rc + r] * la[(size_t)r * Kk + k];
  we[i] = (bf16_t)acc;
}

// ---------------------------------------------------------------------------
// 2) RMSNorm, bf16 output. One 256-thread block per row.
// ---------------------------------------------------------------------------
__global__ __launch_bounds__(256) void rmsnorm_bf16(
    const float* __restrict__ x, const float* __restrict__ w,
    bf16_t* __restrict__ out, int Dn)
{
  __shared__ float red[256];
  size_t row = blockIdx.x;
  const float* xr = x + row * (size_t)Dn;
  float ss = 0.f;
  for (int i = threadIdx.x; i < Dn; i += 256) { float v = xr[i]; ss += v * v; }
  red[threadIdx.x] = ss;
  __syncthreads();
  for (int s = 128; s > 0; s >>= 1) {
    if ((int)threadIdx.x < s) red[threadIdx.x] += red[threadIdx.x + s];
    __syncthreads();
  }
  float rstd = rsqrtf(red[0] / (float)Dn + 1e-5f);
  for (int i = threadIdx.x; i < Dn; i += 256)
    out[row * (size_t)Dn + i] = (bf16_t)(xr[i] * rstd * w[i]);
}

// ---------------------------------------------------------------------------
// 3) WMMA GEMM:  C[M,N] = A[M,K](bf16) · W[N,K]^T(bf16)
//    Compile-time epilogue: (+bias) (*silu(gate)) (+res), f32 and/or bf16 out.
//    8 waves / block, 128x128 block tile, 32x64 per-wave tile.
//    Requires M%128==0, N%128==0, K%32==0 (true for all uses here).
// ---------------------------------------------------------------------------
template<bool HB, bool HG, bool HR, bool OF, bool OB>
__global__ __launch_bounds__(256) void gemm_bf16_wmma(
    const bf16_t* __restrict__ A, const bf16_t* __restrict__ W,
    const float*  __restrict__ bias, const float* __restrict__ res,
    const bf16_t* __restrict__ gate,
    float* __restrict__ Cf, bf16_t* __restrict__ Cb,
    int M, int N, int K)
{
  const int lane  = threadIdx.x & 31;
  const int wave  = threadIdx.x >> 5;
  const int m0    = blockIdx.y * 128 + (wave >> 1) * 32;
  const int n0    = blockIdx.x * 128 + (wave & 1) * 64;
  const int lhalf = lane >> 4;   // 0 or 1
  const int lmod  = lane & 15;
  const int koff  = lhalf * 16;

  v8f acc[2][4];
  #pragma unroll
  for (int mi = 0; mi < 2; ++mi)
    #pragma unroll
    for (int nj = 0; nj < 4; ++nj)
      #pragma unroll
      for (int v = 0; v < 8; ++v) acc[mi][nj][v] = 0.f;

  for (int k0 = 0; k0 < K; k0 += 32) {
    // A fragments (16x32 bf16, per-lane contiguous K chunks {0..15}/{16..31})
    v16bf af[2];
    #pragma unroll
    for (int mi = 0; mi < 2; ++mi) {
      const bf16_t* ap = A + (size_t)(m0 + mi * 16 + lmod) * K + k0;
      v8bf lo = *(const v8bf*)(ap + koff);
      v8bf hi = *(const v8bf*)(ap + 8 + koff);
      #pragma unroll
      for (int i = 0; i < 8; ++i) { af[mi][i] = lo[i]; af[mi][8 + i] = hi[i]; }
    }
    // B fragments (32x16 bf16): lane = column n, 16 contiguous K from weight row
    v16bf bfr[4];
    #pragma unroll
    for (int nj = 0; nj < 4; ++nj) {
      const bf16_t* wp = W + (size_t)(n0 + nj * 16 + lmod) * K + k0 + lhalf * 16;
      bfr[nj] = *(const v16bf*)wp;
    }
    #pragma unroll
    for (int mi = 0; mi < 2; ++mi)
      #pragma unroll
      for (int nj = 0; nj < 4; ++nj)
        acc[mi][nj] = WMMA_BF16(af[mi], bfr[nj], acc[mi][nj]);
  }

  // Epilogue. C layout: element v of v8f -> row m0+mi*16+8*lhalf+v, col n0+nj*16+lmod
  #pragma unroll
  for (int mi = 0; mi < 2; ++mi)
    #pragma unroll
    for (int nj = 0; nj < 4; ++nj) {
      const int col = n0 + nj * 16 + lmod;
      const float bcol = HB ? bias[col] : 0.f;
      #pragma unroll
      for (int v = 0; v < 8; ++v) {
        int row = m0 + mi * 16 + lhalf * 8 + v;
        size_t idx = (size_t)row * N + col;
        float val = acc[mi][nj][v];
        if (HB) val += bcol;
        if (HG) {
          float g = (float)gate[idx];
          // silu(g) = g * sigmoid(g); v_rcp_f32 is plenty for an activation
          val *= g * __builtin_amdgcn_rcpf(1.f + __expf(-g));
        }
        if (HR) val += res[idx];
        if (OF) Cf[idx] = val;
        if (OB) Cb[idx] = (bf16_t)val;
      }
    }
}

// ---------------------------------------------------------------------------
// 4) RoPE + [B,S,H,HD]->[B,H,S,HD] relayout (bf16), and V->[B,H,HD,S]
// ---------------------------------------------------------------------------
__global__ __launch_bounds__(256) void rope_transpose(
    const bf16_t* __restrict__ in, const float* __restrict__ cosb,
    const float* __restrict__ sinb, bf16_t* __restrict__ out)
{
  size_t i = (size_t)blockIdx.x * 256 + threadIdx.x;
  const size_t total = (size_t)Bc * Sc * Hc * HDc;
  if (i >= total) return;
  int d = (int)(i % HDc);
  int h = (int)((i / HDc) % Hc);
  int s = (int)((i / ((size_t)HDc * Hc)) % Sc);
  int b = (int)(i / ((size_t)HDc * Hc * Sc));
  float xv = (float)in[i];
  float ov = (float)in[i + (d < HDc / 2 ? HDc / 2 : -(HDc / 2))];
  float c  = cosb[(size_t)s * HDc + d];
  float sn = sinb[(size_t)s * HDc + d];
  float rot = (d < HDc / 2) ? -ov : ov;
  out[(((size_t)b * Hc + h) * Sc + s) * HDc + d] = (bf16_t)(xv * c + rot * sn);
}

__global__ __launch_bounds__(256) void v_transpose(
    const bf16_t* __restrict__ in, bf16_t* __restrict__ out)
{
  size_t i = (size_t)blockIdx.x * 256 + threadIdx.x;
  const size_t total = (size_t)Bc * Sc * Hc * HDc;
  if (i >= total) return;
  int d = (int)(i % HDc);
  int h = (int)((i / HDc) % Hc);
  int s = (int)((i / ((size_t)HDc * Hc)) % Sc);
  int b = (int)(i / ((size_t)HDc * Hc * Sc));
  out[(((size_t)b * Hc + h) * HDc + d) * Sc + s] = in[i];
}

// ---------------------------------------------------------------------------
// 5) Flash attention. One wave = one 16-query tile; 32-key blocks.
//    Q,K: [B,H,S,HD] bf16; V: [B,H,HD,S] bf16; O: [B,S,H,HD] bf16.
// ---------------------------------------------------------------------------
__global__ __launch_bounds__(256) void attn_fwd(
    const bf16_t* __restrict__ Q, const bf16_t* __restrict__ Km,
    const bf16_t* __restrict__ Vt, const int* __restrict__ am,
    bf16_t* __restrict__ O)
{
  __shared__ __align__(16) bf16_t pst_all[8][16][40]; // row stride 80B (16B-aligned)
  const int lane  = threadIdx.x & 31;
  const int wave  = threadIdx.x >> 5;
  const int lhalf = lane >> 4;
  const int lmod  = lane & 15;
  const int koff  = lhalf * 16;

  const int qblocks = Sc / 16;
  int idx  = blockIdx.x * 8 + wave;       // over (b,h,sblk)
  int sblk = idx % qblocks;
  int bh   = idx / qblocks;
  int b    = bh / Hc;
  int h    = bh % Hc;
  int q0   = sblk * 16;

  const bf16_t* Qb = Q  + (size_t)bh * Sc * HDc;
  const bf16_t* Kb = Km + (size_t)bh * Sc * HDc;
  const bf16_t* Vb = Vt + (size_t)bh * HDc * Sc;
  bf16_t (*pst)[40] = pst_all[wave];

  // Q A-fragments for the 4 K-steps of HD=128
  v16bf qf[4];
  {
    const bf16_t* qr = Qb + (size_t)(q0 + lmod) * HDc;
    #pragma unroll
    for (int kk = 0; kk < 4; ++kk) {
      v8bf lo = *(const v8bf*)(qr + kk * 32 + koff);
      v8bf hi = *(const v8bf*)(qr + kk * 32 + 8 + koff);
      #pragma unroll
      for (int i = 0; i < 8; ++i) { qf[kk][i] = lo[i]; qf[kk][8 + i] = hi[i]; }
    }
  }

  v8f oacc[8];
  #pragma unroll
  for (int dt = 0; dt < 8; ++dt)
    #pragma unroll
    for (int v = 0; v < 8; ++v) oacc[dt][v] = 0.f;
  float mrun[8], lrun[8];
  #pragma unroll
  for (int v = 0; v < 8; ++v) { mrun[v] = -1e30f; lrun[v] = 0.f; }
  const float scale = 0.088388347648318447f; // 1/sqrt(128)
  const int mbase = q0 + lhalf * 8;

  for (int kb = 0; kb <= q0 + 15; kb += 32) {
    // S = Q·K^T for 32 keys (two 16-col halves)
    v8f sc[2];
    #pragma unroll
    for (int nh = 0; nh < 2; ++nh)
      #pragma unroll
      for (int v = 0; v < 8; ++v) sc[nh][v] = 0.f;
    #pragma unroll
    for (int nh = 0; nh < 2; ++nh) {
      const bf16_t* kr = Kb + (size_t)(kb + nh * 16 + lmod) * HDc + lhalf * 16;
      #pragma unroll
      for (int kk = 0; kk < 4; ++kk) {
        v16bf bfrag = *(const v16bf*)(kr + kk * 32);
        sc[nh] = WMMA_BF16(qf[kk], bfrag, sc[nh]);
      }
    }
    // scale + causal mask + key mask
    #pragma unroll
    for (int nh = 0; nh < 2; ++nh) {
      int n = kb + nh * 16 + lmod;
      int kok = am[(size_t)b * Sc + n];
      #pragma unroll
      for (int v = 0; v < 8; ++v) {
        float val = sc[nh][v] * scale;
        sc[nh][v] = (kok != 0 && n <= mbase + v) ? val : -1e30f;
      }
    }
    // row max (rows live in one 16-lane half; masks 1/2/4/8 stay inside it)
    float mx[8];
    #pragma unroll
    for (int v = 0; v < 8; ++v) mx[v] = fmaxf(sc[0][v], sc[1][v]);
    #pragma unroll
    for (int off = 1; off < 16; off <<= 1)
      #pragma unroll
      for (int v = 0; v < 8; ++v) mx[v] = fmaxf(mx[v], __shfl_xor(mx[v], off, 32));
    float fac[8];
    #pragma unroll
    for (int v = 0; v < 8; ++v) {
      float mn = fmaxf(mrun[v], mx[v]);
      fac[v] = __expf(mrun[v] - mn);
      mrun[v] = mn;
    }
    float rs[8];
    #pragma unroll
    for (int v = 0; v < 8; ++v) rs[v] = 0.f;
    #pragma unroll
    for (int nh = 0; nh < 2; ++nh)
      #pragma unroll
      for (int v = 0; v < 8; ++v) {
        float p = __expf(sc[nh][v] - mrun[v]);
        sc[nh][v] = p; rs[v] += p;
      }
    #pragma unroll
    for (int off = 1; off < 16; off <<= 1)
      #pragma unroll
      for (int v = 0; v < 8; ++v) rs[v] += __shfl_xor(rs[v], off, 32);
    #pragma unroll
    for (int v = 0; v < 8; ++v) lrun[v] = lrun[v] * fac[v] + rs[v];
    #pragma unroll
    for (int dt = 0; dt < 8; ++dt)
      #pragma unroll
      for (int v = 0; v < 8; ++v) oacc[dt][v] *= fac[v];

    // P: C-layout -> A-layout via LDS (same-wave LDS ops are in order)
    {
      int mrow = lhalf * 8;
      #pragma unroll
      for (int v = 0; v < 8; ++v) {
        pst[mrow + v][lmod]      = (bf16_t)sc[0][v];
        pst[mrow + v][lmod + 16] = (bf16_t)sc[1][v];
      }
    }
    v16bf pf;
    {
      v8bf lo = *(const v8bf*)(&pst[lmod][koff]);
      v8bf hi = *(const v8bf*)(&pst[lmod][8 + koff]);
      #pragma unroll
      for (int i = 0; i < 8; ++i) { pf[i] = lo[i]; pf[8 + i] = hi[i]; }
    }
    // O += P · V  (V transposed => contiguous 16-key B fragments)
    #pragma unroll
    for (int dt = 0; dt < 8; ++dt) {
      const bf16_t* vr = Vb + (size_t)(dt * 16 + lmod) * Sc + kb + lhalf * 16;
      v16bf vfrag = *(const v16bf*)vr;
      oacc[dt] = WMMA_BF16(pf, vfrag, oacc[dt]);
    }
  }

  // normalize and write O in [B,S,H,HD]
  #pragma unroll
  for (int v = 0; v < 8; ++v) {
    int srow = q0 + lhalf * 8 + v;
    int qok = am[(size_t)b * Sc + srow];
    float inv = (qok != 0) ? __builtin_amdgcn_rcpf(fmaxf(lrun[v], 1e-20f)) : 0.f;
    #pragma unroll
    for (int dt = 0; dt < 8; ++dt) {
      float val = oacc[dt][v] * inv;
      O[(((size_t)b * Sc + srow) * Hc + h) * HDc + dt * 16 + lmod] = (bf16_t)val;
    }
  }
}

// ---------------------------------------------------------------------------
// Launch
// ---------------------------------------------------------------------------
extern "C" void kernel_launch(void* const* d_in, const int* in_sizes, int n_in,
                              void* d_out, int out_size, void* d_ws, size_t ws_size,
                              hipStream_t stream) {
  (void)in_sizes; (void)n_in; (void)out_size; (void)ws_size;
  const float* x   = (const float*)d_in[0];
  const float* nw1 = (const float*)d_in[1];
  const float* nw2 = (const float*)d_in[2];
  const float *W[7], *Bi[7], *LA[7], *LB[7];
  for (int i = 0; i < 7; ++i) {
    W[i]  = (const float*)d_in[3 + i * 4];
    Bi[i] = (const float*)d_in[4 + i * 4];
    LA[i] = (const float*)d_in[5 + i * 4];
    LB[i] = (const float*)d_in[6 + i * 4];
  }
  const float* cosb = (const float*)d_in[31];
  const float* sinb = (const float*)d_in[32];
  const int*   am   = (const int*)d_in[33];
  float* outp = (float*)d_out;

  char* ws = (char*)d_ws;
  size_t off = 0;
  auto alloc = [&](size_t bytes) -> void* {
    off = (off + 255) & ~(size_t)255;
    void* p = ws + off;
    off += bytes;
    return p;
  };

  const size_t MD = (size_t)Mc * Dc, MF = (size_t)Mc * FFc;
  const int wN[7] = {Dc, Dc, Dc, Dc, FFc, FFc, Dc};
  const int wK[7] = {Dc, Dc, Dc, Dc, Dc, Dc, FFc};
  bf16_t* we[7];
  for (int i = 0; i < 7; ++i) we[i] = (bf16_t*)alloc((size_t)wN[i] * wK[i] * 2);
  bf16_t* xn1   = (bf16_t*)alloc(MD * 2);
  bf16_t* qlin  = (bf16_t*)alloc(MD * 2);
  bf16_t* klin  = (bf16_t*)alloc(MD * 2);
  bf16_t* vlin  = (bf16_t*)alloc(MD * 2);
  bf16_t* qb    = (bf16_t*)alloc(MD * 2);
  bf16_t* kbf   = (bf16_t*)alloc(MD * 2);
  bf16_t* vt    = (bf16_t*)alloc(MD * 2);
  bf16_t* attnb = (bf16_t*)alloc(MD * 2);
  float*  xmed  = (float*) alloc(MD * 4);
  bf16_t* xn2   = (bf16_t*)alloc(MD * 2);
  bf16_t* gb    = (bf16_t*)alloc(MF * 2);
  bf16_t* hb    = (bf16_t*)alloc(MF * 2);

  dim3 blk(256);
  // 1) fold LoRA into bf16 effective weights
  for (int i = 0; i < 7; ++i) {
    size_t total = (size_t)wN[i] * wK[i];
    lora_fold<<<dim3((unsigned)((total + 255) / 256)), blk, 0, stream>>>(
        W[i], LA[i], LB[i], we[i], wN[i], wK[i]);
  }
  // 2) rmsnorm 1
  rmsnorm_bf16<<<dim3(Mc), blk, 0, stream>>>(x, nw1, xn1, Dc);
  // 3) q,k,v projections (bias, bf16 out)
  dim3 gD(Dc / 128, Mc / 128), gF(FFc / 128, Mc / 128);
  gemm_bf16_wmma<true,false,false,false,true><<<gD, blk, 0, stream>>>(
      xn1, we[0], Bi[0], nullptr, nullptr, nullptr, qlin, Mc, Dc, Dc);
  gemm_bf16_wmma<true,false,false,false,true><<<gD, blk, 0, stream>>>(
      xn1, we[1], Bi[1], nullptr, nullptr, nullptr, klin, Mc, Dc, Dc);
  gemm_bf16_wmma<true,false,false,false,true><<<gD, blk, 0, stream>>>(
      xn1, we[2], Bi[2], nullptr, nullptr, nullptr, vlin, Mc, Dc, Dc);
  // 4) rope + relayouts
  unsigned eb = (unsigned)((MD + 255) / 256);
  rope_transpose<<<dim3(eb), blk, 0, stream>>>(qlin, cosb, sinb, qb);
  rope_transpose<<<dim3(eb), blk, 0, stream>>>(klin, cosb, sinb, kbf);
  v_transpose  <<<dim3(eb), blk, 0, stream>>>(vlin, vt);
  // 5) flash attention
  attn_fwd<<<dim3(Bc * Hc * (Sc / 16) / 8), blk, 0, stream>>>(qb, kbf, vt, am, attnb);
  // 6) o-proj + residual (bias, res, f32 out)
  gemm_bf16_wmma<true,false,true,true,false><<<gD, blk, 0, stream>>>(
      attnb, we[3], Bi[3], x, nullptr, xmed, nullptr, Mc, Dc, Dc);
  // 7) rmsnorm 2
  rmsnorm_bf16<<<dim3(Mc), blk, 0, stream>>>(xmed, nw2, xn2, Dc);
  // 8) gate, then up with fused silu(gate) multiply
  gemm_bf16_wmma<true,false,false,false,true><<<gF, blk, 0, stream>>>(
      xn2, we[4], Bi[4], nullptr, nullptr, nullptr, gb, Mc, FFc, Dc);
  gemm_bf16_wmma<true,true,false,false,true><<<gF, blk, 0, stream>>>(
      xn2, we[5], Bi[5], nullptr, gb, nullptr, hb, Mc, FFc, Dc);
  // 9) down-proj + residual -> output (bias, res, f32 out)
  gemm_bf16_wmma<true,false,true,true,false><<<gD, blk, 0, stream>>>(
      hb, we[6], Bi[6], xmed, nullptr, outp, nullptr, Mc, Dc, FFc);
}
